// TrajectoryLSTM_19774029431074
// MI455X (gfx1250) — matrix operand order repaired and checked by reading, not compiled, not run on hardware
//
#include <hip/hip_runtime.h>

// ---------------------------------------------------------------------------
// Fused seq2seq LSTM for MI455X (gfx1250, wave32, WMMA + TDM).
// ~0.84 TFLOP of K=128 GEMMs with a serial time recurrence -> fuse everything,
// keep weights WGP-resident (LDS via TENSOR_LOAD_TO_LDS + VGPRs), h/c state in
// LDS/VGPRs. Only input/output and a ~1MB packed-weight workspace touch HBM.
// ---------------------------------------------------------------------------

typedef __attribute__((ext_vector_type(16))) _Float16 v16h;
typedef __attribute__((ext_vector_type(8)))  _Float16 v8h;
typedef __attribute__((ext_vector_type(8)))  float    v8f;
typedef unsigned int uint32x4 __attribute__((ext_vector_type(4)));
typedef int          int32x8  __attribute__((ext_vector_type(8)));
typedef int          int32x4  __attribute__((ext_vector_type(4)));

#define HDIM   128
#define GDIM   512                 // 4*H gate width
#define TIN    20
#define TDEC   45
#define MTILE  32                  // batch rows per workgroup (2 M-tiles)
#define HPAD   (HDIM + 8)          // f16 row pitch to de-conflict LDS banks
#define MAT_ELEMS 65536            // f16 elements per packed 512x128 matrix

// dynamic LDS layout (bytes)
#define SMEM_W0    0                                   // layer-0 recurrent weights (packed frags)
#define SMEM_W1    131072                              // layer-1 input weights (packed frags)
#define SMEM_H0    262144                              // h0 double buffer, f16 [2][MTILE][HPAD]
#define SMEM_H1    (SMEM_H0 + 2*MTILE*HPAD*2)
#define SMEM_X     (SMEM_H1 + 2*MTILE*HPAD*2)          // x tile f32 [MTILE][2]
#define SMEM_FC    (SMEM_X + MTILE*2*4)                // fc_W f32 [2*HDIM]
#define SMEM_TOTAL (SMEM_FC + 2*HDIM*4)

typedef _Float16 hrow_t[HPAD];

#define WMMA(a, b, c) \
    __builtin_amdgcn_wmma_f32_16x16x32_f16(false, (a), false, (b), (short)0, (c), false, false)

// ---------------------------------------------------------------------------
// Prologue: swizzle a 512x128 f32 weight matrix into WMMA B-fragment order.
// k order used here (same as loadA): lanes 0-15 hold column n with K=0..15 contiguous,
// lanes 16-31 hold column n with K=16..31. Packed as [nt][kt][lane][16] f16 so
// each lane's fragment is a contiguous 32B load.
// ---------------------------------------------------------------------------
__global__ void pack_weights_f16(const float* __restrict__ W, _Float16* __restrict__ dst) {
  int tid = blockIdx.x * blockDim.x + threadIdx.x;
  if (tid >= MAT_ELEMS) return;
  int e    = tid & 15;
  int lane = (tid >> 4) & 31;
  int kt   = (tid >> 9) & 3;
  int nt   = tid >> 11;                       // 0..31
  int n = nt * 16 + (lane & 15);              // gate column
  int k = kt * 32 + ((lane < 16) ? e : 16 + e);
  dst[tid] = (_Float16)W[n * HDIM + k];
}

__global__ void pack_bias_sum(const float* __restrict__ a, const float* __restrict__ b,
                              float* __restrict__ dst) {
  int i = blockIdx.x * blockDim.x + threadIdx.x;
  if (i < GDIM) dst[i] = a[i] + b[i];
}

// ---------------------------------------------------------------------------
// Main fused kernel: 8 waves per WG; wave w owns h-columns [16w,16w+16) and the
// corresponding i/f/g/o gate tiles (nt = w, w+8, w+16, w+24) so the LSTM
// elementwise update happens entirely in the wave's accumulator registers.
// ---------------------------------------------------------------------------
__global__ __launch_bounds__(256, 1)
void seq2seq_lstm(const _Float16* __restrict__ wsf, const float* __restrict__ wsb,
                  const float* __restrict__ xin,  const float* __restrict__ encWx,
                  const float* __restrict__ decWx, const float* __restrict__ fcW,
                  const float* __restrict__ fcB,  float* __restrict__ out) {
  extern __shared__ char smem[];
  _Float16* ldsW0 = (_Float16*)(smem + SMEM_W0);
  _Float16* ldsW1 = (_Float16*)(smem + SMEM_W1);
  hrow_t*   sh0   = (hrow_t*)(smem + SMEM_H0);       // [2*MTILE] rows
  hrow_t*   sh1   = (hrow_t*)(smem + SMEM_H1);
  float (*sx)[2]  = (float (*)[2])(smem + SMEM_X);
  float*    sfc   = (float*)(smem + SMEM_FC);

  const int tid  = threadIdx.x;
  const int lane = tid & 31;
  const int w    = tid >> 5;          // wave id 0..7
  const int lrow = lane & 15;
  const int lhi  = lane >> 4;         // 0: lanes 0-15, 1: lanes 16-31
  const int hcol = (w << 4) + lrow;   // this lane's h column
  const long b0  = (long)blockIdx.x * MTILE;

  v16h  Wc[4][4];                     // register-resident layer-1 recurrent weights
  float bg0[4], bg1[4], wx[4][2];     // gate biases; K=2 input-projection weights
  v8f   c0[2], c1[2];                 // cell state: [mt], element e -> row mt*16+lhi*8+e

  auto zero8 = []() { v8f z;
    #pragma unroll
    for (int i = 0; i < 8; ++i) z[i] = 0.f;
    return z; };
  auto sigm = [](float x) { return 1.0f / (1.0f + __expf(-x)); };
  auto ftanh = [](float x) {
    x = fminf(fmaxf(x, -15.f), 15.f);
    float e = __expf(2.f * x);
    return (e - 1.f) / (e + 1.f); };

  // A-fragment (this code's k order, same as packed B): lane<16 -> K 0..15, lane>=16 -> K 16..31
  auto loadA = [&](const hrow_t* sh, int mt, int kt) -> v16h {
    const _Float16* p = &sh[mt * 16 + lrow][kt * 32 + lhi * 16];
    v8h lo = *(const v8h*)p;
    v8h hi = *(const v8h*)(p + 8);
    v16h a;
    #pragma unroll
    for (int i = 0; i < 8; ++i) { a[i] = lo[i]; a[i + 8] = hi[i]; }
    return a; };

  auto loadFrag = [&](const _Float16* p) -> v16h {
    v8h lo = *(const v8h*)p;
    v8h hi = *(const v8h*)(p + 8);
    v16h a;
    #pragma unroll
    for (int i = 0; i < 8; ++i) { a[i] = lo[i]; a[i + 8] = hi[i]; }
    return a; };

  auto loadB = [&](const _Float16* wb, int g, int kt) -> v16h {
    return loadFrag(wb + ((((g * 8 + w) * 4 + kt) * 32 + lane) * 16)); };

  // -------------------------------------------------------------------------
  // TDM: DMA one packed 128KB weight matrix (16384 x 8B elements, 1 row tile)
  // from global into LDS. Descriptor per CDNA5 ISA ch.8 (groups 2/3 zero).
  // Issued by a single wave; completion via TENSORcnt. This toolchain's
  // builtin takes 6 args: (g0 x4, g1 x8, grp2 x4, grp3 x4, extra x8, cpol).
  // -------------------------------------------------------------------------
  auto tdm_stage = [&](unsigned lds_byte, const _Float16* gsrc) {
#if __has_builtin(__builtin_amdgcn_tensor_load_to_lds)
    unsigned long long ga = (unsigned long long)gsrc;
    uint32x4 g0;
    g0[0] = 1u;                                             // count=1 valid descriptor
    g0[1] = lds_byte;                                       // lds_addr
    g0[2] = (unsigned)(ga & 0xffffffffu);                   // global_addr[31:0]
    g0[3] = ((unsigned)(ga >> 32) & 0x01ffffffu) | (2u << 30); // addr[56:32] | type=2
    int32x8 g1;
    g1[0] = 0x00030000;            // workgroup_mask=0, data_size=3 (8B), no pad/iter
    g1[1] = 0x40000000;            // tensor_dim0[15:0]=16384 at bits 63:48
    g1[2] = 0x00010000;            // tensor_dim1=1 at bits 111:80
    g1[3] = 0x40000000;            // tile_dim0=16384 at bits 127:112
    g1[4] = 0x00000001;            // tile_dim1=1, tile_dim2=0
    g1[5] = 16384;                 // tensor_dim0_stride[31:0]
    g1[6] = 0x40000000;            // tensor_dim1_stride[15:0]=16384 at bits 223:208
    g1[7] = 0;
    int32x4 z4 = {0, 0, 0, 0};
    int32x8 z8 = {0, 0, 0, 0, 0, 0, 0, 0};
    __builtin_amdgcn_tensor_load_to_lds(g0, g1, z4, z4, z8, 0);
#else
    const uint4* s = (const uint4*)gsrc;
    uint4* d = (uint4*)(smem + lds_byte);
    for (int i = lane; i < MAT_ELEMS / 8; i += 32) d[i] = s[i];
#endif
  };

  // Stage a phase's weights: two matrices -> LDS (TDM), one -> VGPRs, biases.
  auto load_phase = [&](int m0, int m1, int m2, int bl0, int bl1, const float* Wx) {
    if (w == 0) {
      tdm_stage(SMEM_W0, wsf + (size_t)m0 * MAT_ELEMS);
    } else if (w == 1) {
      tdm_stage(SMEM_W1, wsf + (size_t)m1 * MAT_ELEMS);
    }
#if __has_builtin(__builtin_amdgcn_tensor_load_to_lds)
    if (w < 2) __builtin_amdgcn_s_wait_tensorcnt(0);
#endif
    const _Float16* rb = wsf + (size_t)m2 * MAT_ELEMS;
    #pragma unroll
    for (int g = 0; g < 4; ++g)
      #pragma unroll
      for (int kt = 0; kt < 4; ++kt)
        Wc[g][kt] = loadFrag(rb + ((((g * 8 + w) * 4 + kt) * 32 + lane) * 16));
    #pragma unroll
    for (int g = 0; g < 4; ++g) {
      int n = g * 128 + hcol;
      bg0[g]   = wsb[bl0 * GDIM + n];
      bg1[g]   = wsb[bl1 * GDIM + n];
      wx[g][0] = Wx[n * 2 + 0];
      wx[g][1] = Wx[n * 2 + 1];
    } };

  // Layer 0 cell: gates = h@Whh^T (WMMA, LDS weights) + x@Wih^T (K=2, VALU) + b
  auto cellA = [&](const hrow_t* shIn, hrow_t* shOut, v8f* c) {
    v8f acc[2][4];
    #pragma unroll
    for (int mt = 0; mt < 2; ++mt)
      #pragma unroll
      for (int g = 0; g < 4; ++g) acc[mt][g] = zero8();
    #pragma unroll
    for (int kt = 0; kt < 4; ++kt) {
      v16h a0 = loadA(shIn, 0, kt);
      v16h a1 = loadA(shIn, 1, kt);
      #pragma unroll
      for (int g = 0; g < 4; ++g) {
        v16h bf = loadB(ldsW0, g, kt);
        acc[0][g] = WMMA(a0, bf, acc[0][g]);
        acc[1][g] = WMMA(a1, bf, acc[1][g]);
      }
    }
    #pragma unroll
    for (int mt = 0; mt < 2; ++mt)
      #pragma unroll
      for (int e = 0; e < 8; ++e) {
        int m = mt * 16 + lhi * 8 + e;
        float xa = sx[m][0], xb = sx[m][1];
        float ig = sigm (acc[mt][0][e] + bg0[0] + wx[0][0] * xa + wx[0][1] * xb);
        float fg = sigm (acc[mt][1][e] + bg0[1] + wx[1][0] * xa + wx[1][1] * xb);
        float gg = ftanh(acc[mt][2][e] + bg0[2] + wx[2][0] * xa + wx[2][1] * xb);
        float og = sigm (acc[mt][3][e] + bg0[3] + wx[3][0] * xa + wx[3][1] * xb);
        float cc = fg * c[mt][e] + ig * gg;
        c[mt][e] = cc;
        shOut[m][hcol] = (_Float16)(og * ftanh(cc));
      } };

  // Layer 1 cell: gates = x@Wih^T (LDS weights) + h@Whh^T (VGPR weights) + b
  auto cellB = [&](const hrow_t* shX, const hrow_t* shR, hrow_t* shOut, v8f* c) {
    v8f acc[2][4];
    #pragma unroll
    for (int mt = 0; mt < 2; ++mt)
      #pragma unroll
      for (int g = 0; g < 4; ++g) acc[mt][g] = zero8();
    #pragma unroll
    for (int kt = 0; kt < 4; ++kt) {
      v16h a0 = loadA(shX, 0, kt);
      v16h a1 = loadA(shX, 1, kt);
      #pragma unroll
      for (int g = 0; g < 4; ++g) {
        v16h bf = loadB(ldsW1, g, kt);
        acc[0][g] = WMMA(a0, bf, acc[0][g]);
        acc[1][g] = WMMA(a1, bf, acc[1][g]);
      }
    }
    #pragma unroll
    for (int kt = 0; kt < 4; ++kt) {
      v16h a0 = loadA(shR, 0, kt);
      v16h a1 = loadA(shR, 1, kt);
      #pragma unroll
      for (int g = 0; g < 4; ++g) {
        acc[0][g] = WMMA(a0, Wc[g][kt], acc[0][g]);
        acc[1][g] = WMMA(a1, Wc[g][kt], acc[1][g]);
      }
    }
    #pragma unroll
    for (int mt = 0; mt < 2; ++mt)
      #pragma unroll
      for (int e = 0; e < 8; ++e) {
        int m = mt * 16 + lhi * 8 + e;
        float ig = sigm (acc[mt][0][e] + bg1[0]);
        float fg = sigm (acc[mt][1][e] + bg1[1]);
        float gg = ftanh(acc[mt][2][e] + bg1[2]);
        float og = sigm (acc[mt][3][e] + bg1[3]);
        float cc = fg * c[mt][e] + ig * gg;
        c[mt][e] = cc;
        shOut[m][hcol] = (_Float16)(og * ftanh(cc));
      } };

  // ---------------- encoder ----------------
  load_phase(0, 1, 2, 0, 1, encWx);
  #pragma unroll
  for (int mt = 0; mt < 2; ++mt) { c0[mt] = zero8(); c1[mt] = zero8(); }
  for (int i = tid; i < 2 * MTILE * HPAD; i += 256) {
    ((_Float16*)sh0)[i] = (_Float16)0.f;
    ((_Float16*)sh1)[i] = (_Float16)0.f;
  }
  if (tid < MTILE) {
    sx[tid][0] = xin[(b0 + tid) * TIN * 2 + 0];
    sx[tid][1] = xin[(b0 + tid) * TIN * 2 + 1];
  }
  __syncthreads();

  int cur = 0;
  #pragma unroll 1
  for (int t = 0; t < TIN; ++t) {
    cellA(sh0 + cur * MTILE, sh0 + (cur ^ 1) * MTILE, c0);
    __syncthreads();
    cellB(sh0 + (cur ^ 1) * MTILE, sh1 + cur * MTILE, sh1 + (cur ^ 1) * MTILE, c1);
    if (t + 1 < TIN && tid < MTILE) {            // prefetch next x tile
      sx[tid][0] = xin[(b0 + tid) * TIN * 2 + (t + 1) * 2 + 0];
      sx[tid][1] = xin[(b0 + tid) * TIN * 2 + (t + 1) * 2 + 1];
    }
    __syncthreads();
    cur ^= 1;
  }

  // ---------------- decoder ----------------
  load_phase(3, 4, 5, 2, 3, decWx);
  if (tid < MTILE) {                              // x0 = input_seq[:, -1, :]
    sx[tid][0] = xin[(b0 + tid) * TIN * 2 + (TIN - 1) * 2 + 0];
    sx[tid][1] = xin[(b0 + tid) * TIN * 2 + (TIN - 1) * 2 + 1];
  }
  if (tid < 2 * HDIM) sfc[tid] = fcW[tid];
  __syncthreads();

  // FC head work distribution: 4 lanes per (m,o) pair, shfl-reduced.
  const int pr   = (w << 3) + (lane >> 2);        // 0..63 -> (m,o)
  const int pm   = pr >> 1;
  const int po   = pr & 1;
  const int part = lane & 3;                      // 32 k-elements per part
  const float fb = fcB[po];

  #pragma unroll 1
  for (int t = 0; t < TDEC; ++t) {
    cellA(sh0 + cur * MTILE, sh0 + (cur ^ 1) * MTILE, c0);
    __syncthreads();
    cellB(sh0 + (cur ^ 1) * MTILE, sh1 + cur * MTILE, sh1 + (cur ^ 1) * MTILE, c1);
    __syncthreads();
    {                                             // pred = h1 @ fc_W^T + fc_b
      const _Float16* hr = sh1[(cur ^ 1) * MTILE + pm] + part * 32;
      const float*    fw = sfc + po * HDIM + part * 32;
      float s = 0.f;
      #pragma unroll
      for (int c8 = 0; c8 < 4; ++c8) {
        v8h hv = *(const v8h*)(hr + c8 * 8);
        #pragma unroll
        for (int j = 0; j < 8; ++j) s += (float)hv[j] * fw[c8 * 8 + j];
      }
      s += __shfl_xor(s, 1, 32);
      s += __shfl_xor(s, 2, 32);
      if (part == 0) {
        s += fb;
        out[(b0 + pm) * TDEC * 2 + t * 2 + po] = s;
        sx[pm][po] = s;                           // feed back as next x
      }
    }
    __syncthreads();
    cur ^= 1;
  }
}

// ---------------------------------------------------------------------------
extern "C" void kernel_launch(void* const* d_in, const int* in_sizes, int n_in,
                              void* d_out, int out_size, void* d_ws, size_t ws_size,
                              hipStream_t stream) {
  const float* input_seq = (const float*)d_in[0];
  const float* enc_Wih0  = (const float*)d_in[1];
  const float* enc_Whh0  = (const float*)d_in[2];
  const float* enc_bih0  = (const float*)d_in[3];
  const float* enc_bhh0  = (const float*)d_in[4];
  const float* enc_Wih1  = (const float*)d_in[5];
  const float* enc_Whh1  = (const float*)d_in[6];
  const float* enc_bih1  = (const float*)d_in[7];
  const float* enc_bhh1  = (const float*)d_in[8];
  const float* dec_Wih0  = (const float*)d_in[9];
  const float* dec_Whh0  = (const float*)d_in[10];
  const float* dec_bih0  = (const float*)d_in[11];
  const float* dec_bhh0  = (const float*)d_in[12];
  const float* dec_Wih1  = (const float*)d_in[13];
  const float* dec_Whh1  = (const float*)d_in[14];
  const float* dec_bih1  = (const float*)d_in[15];
  const float* dec_bhh1  = (const float*)d_in[16];
  const float* fc_W      = (const float*)d_in[17];
  const float* fc_b      = (const float*)d_in[18];

  _Float16* wsf = (_Float16*)d_ws;                           // 6 packed matrices
  float*    wsb = (float*)((char*)d_ws + 6 * MAT_ELEMS * 2); // 4 bias sums

  const int PB = (MAT_ELEMS + 255) / 256;
  pack_weights_f16<<<PB, 256, 0, stream>>>(enc_Whh0, wsf + 0 * MAT_ELEMS);
  pack_weights_f16<<<PB, 256, 0, stream>>>(enc_Wih1, wsf + 1 * MAT_ELEMS);
  pack_weights_f16<<<PB, 256, 0, stream>>>(enc_Whh1, wsf + 2 * MAT_ELEMS);
  pack_weights_f16<<<PB, 256, 0, stream>>>(dec_Whh0, wsf + 3 * MAT_ELEMS);
  pack_weights_f16<<<PB, 256, 0, stream>>>(dec_Wih1, wsf + 4 * MAT_ELEMS);
  pack_weights_f16<<<PB, 256, 0, stream>>>(dec_Whh1, wsf + 5 * MAT_ELEMS);
  pack_bias_sum<<<2, 256, 0, stream>>>(enc_bih0, enc_bhh0, wsb + 0 * GDIM);
  pack_bias_sum<<<2, 256, 0, stream>>>(enc_bih1, enc_bhh1, wsb + 1 * GDIM);
  pack_bias_sum<<<2, 256, 0, stream>>>(dec_bih0, dec_bhh0, wsb + 2 * GDIM);
  pack_bias_sum<<<2, 256, 0, stream>>>(dec_bih1, dec_bhh1, wsb + 3 * GDIM);

  int Bsz  = in_sizes[0] / (TIN * 2);
  int grid = Bsz / MTILE;
  seq2seq_lstm<<<grid, 256, SMEM_TOTAL, stream>>>(
      wsf, wsb, input_seq, enc_Wih0, dec_Wih0, fc_W, fc_b, (float*)d_out);
}
